// BlockCrossAttention_44787918962965
// MI455X (gfx1250) — matrix operand order repaired and checked by reading, not compiled, not run on hardware
//
#include <hip/hip_runtime.h>

typedef unsigned short u16;
typedef unsigned int   u32;
typedef __attribute__((ext_vector_type(16))) __bf16 bf16x16;
typedef __attribute__((ext_vector_type(8)))  float  f32x8;
typedef __attribute__((ext_vector_type(4)))  u32    u32x4;
typedef __attribute__((ext_vector_type(4)))  float  f32x4;
typedef __attribute__((ext_vector_type(4)))  u16    u16x4;

#define BB   4
#define HH   64
#define SS   64
#define LL   4
#define DD   1024
#define NHH  16
#define DHH  64
#define LOG_H 4.1588830833596715f   /* ln(64) */

__device__ __forceinline__ u16 f2bf(float x) {
  u32 u = __float_as_uint(x);
  u = u + 0x7FFFu + ((u >> 16) & 1u);      // round-to-nearest-even
  return (u16)(u >> 16);
}

union Frag { bf16x16 v; u32x4 q[2]; };

// ---------------------------------------------------------------------------
// C[M,N] = A_bf16[M,K] @ W_bf16[N,K]^T + bias_scale * bias[N]
// One wave computes a 16x64 strip: A fragment loaded once per K-step and
// reused across 4 B fragments -> 4x v_wmma_f32_16x16x32_bf16 per K-step.
// ---------------------------------------------------------------------------
__global__ __launch_bounds__(32)
void gemm_bf16_wmma(const u16* __restrict__ A, const u16* __restrict__ W,
                    const float* __restrict__ bias, float bias_scale,
                    float* __restrict__ C, int M, int N, int K) {
  const int n_base = blockIdx.x * 64;
  const int m_base = blockIdx.y * 16;
  const int lane = threadIdx.x;        // 0..31, wave32
  const int g  = lane >> 4;            // lane half
  const int mr = lane & 15;

  const u16* arow = A + (size_t)(m_base + mr) * K;   // A row m, K contiguous
  const u16* wrow0 = W + (size_t)(n_base + mr) * K;  // W row n == B column n
  const u16* wrow1 = wrow0 + (size_t)16 * K;
  const u16* wrow2 = wrow0 + (size_t)32 * K;
  const u16* wrow3 = wrow0 + (size_t)48 * K;

  f32x8 acc0 = {}, acc1 = {}, acc2 = {}, acc3 = {};
#pragma unroll 2
  for (int k0 = 0; k0 < K; k0 += 32) {
    Frag a, b0, b1, b2, b3;
    // A 16x32 bf16, k order used here (same for B): half-lane g holds 16 contiguous K starting at g*16
    a.q[0] = *reinterpret_cast<const u32x4*>(arow + k0 + g * 16);
    a.q[1] = *reinterpret_cast<const u32x4*>(arow + k0 + g * 16 + 8);
    // B 32x16 bf16, k order used here (same as A): half-lane g holds 16 contiguous K starting at g*16
    b0.q[0] = *reinterpret_cast<const u32x4*>(wrow0 + k0 + g * 16);
    b0.q[1] = *reinterpret_cast<const u32x4*>(wrow0 + k0 + g * 16 + 8);
    b1.q[0] = *reinterpret_cast<const u32x4*>(wrow1 + k0 + g * 16);
    b1.q[1] = *reinterpret_cast<const u32x4*>(wrow1 + k0 + g * 16 + 8);
    b2.q[0] = *reinterpret_cast<const u32x4*>(wrow2 + k0 + g * 16);
    b2.q[1] = *reinterpret_cast<const u32x4*>(wrow2 + k0 + g * 16 + 8);
    b3.q[0] = *reinterpret_cast<const u32x4*>(wrow3 + k0 + g * 16);
    b3.q[1] = *reinterpret_cast<const u32x4*>(wrow3 + k0 + g * 16 + 8);
    acc0 = __builtin_amdgcn_wmma_f32_16x16x32_bf16(
        false, a.v, false, b0.v, (short)0, acc0, false, false);
    acc1 = __builtin_amdgcn_wmma_f32_16x16x32_bf16(
        false, a.v, false, b1.v, (short)0, acc1, false, false);
    acc2 = __builtin_amdgcn_wmma_f32_16x16x32_bf16(
        false, a.v, false, b2.v, (short)0, acc2, false, false);
    acc3 = __builtin_amdgcn_wmma_f32_16x16x32_bf16(
        false, a.v, false, b3.v, (short)0, acc3, false, false);
  }

  const float bb0 = bias[n_base + mr] * bias_scale;
  const float bb1 = bias[n_base + 16 + mr] * bias_scale;
  const float bb2 = bias[n_base + 32 + mr] * bias_scale;
  const float bb3 = bias[n_base + 48 + mr] * bias_scale;
#pragma unroll
  for (int r = 0; r < 8; ++r) {
    // C/D layout: VGPR r -> row M = r + 8*g, col N = lane%16
    float* crow = C + (size_t)(m_base + r + 8 * g) * N + n_base + mr;
    crow[0]  = acc0[r] + bb0;
    crow[16] = acc1[r] + bb1;
    crow[32] = acc2[r] + bb2;
    crow[48] = acc3[r] + bb3;
  }
}

// ---------------------------------------------------------------------------
// Mean-pool k over S for every (b,h,l); emit bf16 rows of km[(b*H+h)*L+l][D].
// ---------------------------------------------------------------------------
__global__ __launch_bounds__(256)
void pool_k_kernel(const float* __restrict__ k, u16* __restrict__ km) {
  const int r  = blockIdx.x;           // (b*H+h)*L + l, 1024 blocks
  const int bh = r >> 2;
  const int l  = r & 3;
  const int t  = threadIdx.x;          // 256 threads * float4 = 1024 floats
  const size_t stride = (size_t)LL * DD;
  const float* p = k + ((size_t)bh * SS * LL + l) * DD + t * 4;
  f32x4 acc = {0.f, 0.f, 0.f, 0.f};
#pragma unroll 4
  for (int s = 0; s < SS; ++s)
    acc += *reinterpret_cast<const f32x4*>(p + (size_t)s * stride);
  const float inv = 1.0f / (float)SS;
  u16x4 o;
  o.x = f2bf(acc.x * inv); o.y = f2bf(acc.y * inv);
  o.z = f2bf(acc.z * inv); o.w = f2bf(acc.w * inv);
  *reinterpret_cast<u16x4*>(km + (size_t)r * DD + t * 4) = o;
}

// ---------------------------------------------------------------------------
// Sum-pool v over S at level L-1 only (the only level the reference uses).
// ---------------------------------------------------------------------------
__global__ __launch_bounds__(256)
void pool_v_kernel(const float* __restrict__ v, u16* __restrict__ vs) {
  const int bh = blockIdx.x;           // 256 blocks
  const int t  = threadIdx.x;
  const size_t stride = (size_t)LL * DD;
  const float* p = v + ((size_t)bh * SS * LL + (LL - 1)) * DD + t * 4;
  f32x4 acc = {0.f, 0.f, 0.f, 0.f};
#pragma unroll 4
  for (int s = 0; s < SS; ++s)
    acc += *reinterpret_cast<const f32x4*>(p + (size_t)s * stride);
  u16x4 o;
  o.x = f2bf(acc.x); o.y = f2bf(acc.y); o.z = f2bf(acc.z); o.w = f2bf(acc.w);
  *reinterpret_cast<u16x4*>(vs + (size_t)bh * DD + t * 4) = o;
}

__global__ void cvt_f32_bf16(const float* __restrict__ in, u16* __restrict__ out, int n) {
  int i = blockIdx.x * blockDim.x + threadIdx.x;
  if (i < n) out[i] = f2bf(in[i]);
}

// ---------------------------------------------------------------------------
// Per (l,b,n): 64x64 scores, softmax over k, entropy per row.
// block = ((l*B + b)*NH + n), 64 threads (one row each).
// ---------------------------------------------------------------------------
__global__ __launch_bounds__(64)
void attn_kernel(const float* __restrict__ qp, const float* __restrict__ kp,
                 float* __restrict__ attn, float* __restrict__ ent) {
  const int blk = blockIdx.x;
  const int n  = blk & 15;
  const int lb = blk >> 4;
  const int b  = lb & 3;
  const int l  = lb >> 2;

  __shared__ float qt[64][65];
  __shared__ float kt[64][65];
  const int q = threadIdx.x;

  const float* qrow = qp + (size_t)(b * HH + q) * DD + n * DHH;
  const float* krow = kp + (size_t)((b * HH + q) * LL + l) * DD + n * DHH;
#pragma unroll 8
  for (int d = 0; d < 64; ++d) { qt[q][d] = qrow[d]; kt[q][d] = krow[d]; }
  __syncthreads();

  float s[64];
  float mx = -3.0e38f;
  for (int kk = 0; kk < 64; ++kk) {
    float acc = 0.f;
#pragma unroll 8
    for (int d = 0; d < 64; ++d) acc += qt[q][d] * kt[kk][d];
    acc *= 0.125f;                    // dh^-0.5
    s[kk] = acc;
    mx = fmaxf(mx, acc);
  }
  float sum = 0.f;
#pragma unroll 8
  for (int kk = 0; kk < 64; ++kk) { s[kk] = __expf(s[kk] - mx); sum += s[kk]; }
  const float inv = 1.0f / sum;
  float e = 0.f;
  float* arow = attn + ((size_t)blk * 64 + q) * 64;
#pragma unroll 4
  for (int kk = 0; kk < 64; ++kk) {
    float p = s[kk] * inv;
    arow[kk] = p;
    e -= p * __logf(p + 1e-8f);
  }
  ent[blk * 64 + q] = e;
}

// ---------------------------------------------------------------------------
// Level gate: entropy mean over heads, /ln(H), softmax(-entropy) over L.
// Single block, thread = (b,q).
// ---------------------------------------------------------------------------
__global__ __launch_bounds__(256)
void wlvl_kernel(const float* __restrict__ ent, float* __restrict__ wl) {
  const int i = threadIdx.x;
  const int b = i >> 6, q = i & 63;
  float e[LL];
#pragma unroll
  for (int l = 0; l < LL; ++l) {
    float s = 0.f;
#pragma unroll
    for (int n = 0; n < NHH; ++n)
      s += ent[(((l * BB + b) * NHH) + n) * 64 + q];
    e[l] = (s / (float)NHH) / LOG_H;
  }
  float mx = -3.0e38f;
#pragma unroll
  for (int l = 0; l < LL; ++l) mx = fmaxf(mx, -e[l]);
  float sum = 0.f, w[LL];
#pragma unroll
  for (int l = 0; l < LL; ++l) { w[l] = __expf(-e[l] - mx); sum += w[l]; }
  const float inv = 1.0f / sum;
#pragma unroll
  for (int l = 0; l < LL; ++l) wl[(l * BB + b) * 64 + q] = w[l] * inv;
}

// ---------------------------------------------------------------------------
// fused = sum_l w_l * attn;  ctx = fused @ v_last;  write ctx with the
// faithful (B,nh,H,dh)->(B,H,nh*dh) row-major scramble, as bf16 for o-proj.
// block = b*NH + n, 64 threads (one q row each).
// ---------------------------------------------------------------------------
__global__ __launch_bounds__(64)
void fuse_kernel(const float* __restrict__ attn, const float* __restrict__ wl,
                 const float* __restrict__ vp, u16* __restrict__ ctx2) {
  const int blk = blockIdx.x;
  const int b = blk >> 4, n = blk & 15;
  __shared__ float vt[64][65];
  const int q = threadIdx.x;

  const float* vrow = vp + (size_t)(b * HH + q) * DD + n * DHH;  // v_last[b][n][k=q][:]
#pragma unroll 8
  for (int d = 0; d < 64; ++d) vt[q][d] = vrow[d];
  __syncthreads();

  float w[LL];
#pragma unroll
  for (int l = 0; l < LL; ++l) w[l] = wl[(l * BB + b) * 64 + q];

  float f[64];
  for (int kk = 0; kk < 64; ++kk) {
    float acc = 0.f;
#pragma unroll
    for (int l = 0; l < LL; ++l)
      acc += w[l] * attn[((size_t)(((l * BB + b) * NHH + n) * 64 + q)) * 64 + kk];
    f[kk] = acc;
  }

  // flat (n,q,d) -> n*4096 + q*64 + d ; row x = n*4 + q/16, col = (q%16)*64 + d
  const int x = n * 4 + (q >> 4);
  const int ybase = (q & 15) * 64;
  u16* orow = ctx2 + (size_t)(b * HH + x) * DD + ybase;
  for (int d = 0; d < 64; ++d) {
    float c = 0.f;
#pragma unroll 8
    for (int kk = 0; kk < 64; ++kk) c += f[kk] * vt[kk][d];
    orow[d] = f2bf(c);
  }
}

// ---------------------------------------------------------------------------
extern "C" void kernel_launch(void* const* d_in, const int* in_sizes, int n_in,
                              void* d_out, int out_size, void* d_ws, size_t ws_size,
                              hipStream_t stream) {
  const float* q  = (const float*)d_in[0];
  const float* k  = (const float*)d_in[1];
  const float* v  = (const float*)d_in[2];
  const float* Wq = (const float*)d_in[3];
  const float* bq = (const float*)d_in[4];
  const float* Wk = (const float*)d_in[5];
  const float* bk = (const float*)d_in[6];
  const float* Wv = (const float*)d_in[7];
  const float* bv = (const float*)d_in[8];
  const float* Wo = (const float*)d_in[9];
  const float* bo = (const float*)d_in[10];
  float* out = (float*)d_out;

  char* w = (char*)d_ws;
  size_t o = 0;
  auto carve = [&](size_t bytes) { char* p = w + o; o += (bytes + 255) & ~(size_t)255; return p; };
  const size_t SZ_W  = (size_t)DD * DD * 2;     // 2 MB, bf16 weight
  const size_t SZ_R  = (size_t)BB * HH * DD;    // 262144 elements

  u16* Wq_b = (u16*)carve(SZ_W);
  u16* Wk_b = (u16*)carve(SZ_W);
  u16* Wv_b = (u16*)carve(SZ_W);
  u16* Wo_b = (u16*)carve(SZ_W);
  u16* q_b  = (u16*)carve(SZ_R * 2);
  u16* km_b = (u16*)carve((size_t)BB * HH * LL * DD * 2);
  u16* vs_b = (u16*)carve(SZ_R * 2);
  float* qp   = (float*)carve(SZ_R * 4);
  float* kp   = (float*)carve((size_t)BB * HH * LL * DD * 4);
  float* vp   = (float*)carve(SZ_R * 4);
  float* attn = (float*)carve((size_t)LL * BB * NHH * 64 * 64 * 4);
  float* ent  = (float*)carve((size_t)LL * BB * NHH * 64 * 4);
  float* wl   = (float*)carve((size_t)LL * BB * 64 * 4);
  u16* ctx2   = (u16*)carve(SZ_R * 2);

  const int NW = DD * DD;
  cvt_f32_bf16<<<(NW + 255) / 256, 256, 0, stream>>>(Wq, Wq_b, NW);
  cvt_f32_bf16<<<(NW + 255) / 256, 256, 0, stream>>>(Wk, Wk_b, NW);
  cvt_f32_bf16<<<(NW + 255) / 256, 256, 0, stream>>>(Wv, Wv_b, NW);
  cvt_f32_bf16<<<(NW + 255) / 256, 256, 0, stream>>>(Wo, Wo_b, NW);
  cvt_f32_bf16<<<((int)SZ_R + 255) / 256, 256, 0, stream>>>(q, q_b, (int)SZ_R);

  pool_k_kernel<<<BB * HH * LL, 256, 0, stream>>>(k, km_b);   // 268 MB HBM stream
  pool_v_kernel<<<BB * HH, 256, 0, stream>>>(v, vs_b);        // 67 MB HBM stream

  // projections (WMMA bf16, f32 accumulate); each wave does a 16x64 strip
  gemm_bf16_wmma<<<dim3(DD / 64, (BB * HH) / 16), 32, 0, stream>>>(
      q_b, Wq_b, bq, 1.0f, qp, BB * HH, DD, DD);
  gemm_bf16_wmma<<<dim3(DD / 64, (BB * HH * LL) / 16), 32, 0, stream>>>(
      km_b, Wk_b, bk, 1.0f, kp, BB * HH * LL, DD, DD);
  gemm_bf16_wmma<<<dim3(DD / 64, (BB * HH) / 16), 32, 0, stream>>>(
      vs_b, Wv_b, bv, (float)SS, vp, BB * HH, DD, DD);

  attn_kernel<<<LL * BB * NHH, 64, 0, stream>>>(qp, kp, attn, ent);
  wlvl_kernel<<<1, BB * 64, 0, stream>>>(ent, wl);
  fuse_kernel<<<BB * NHH, 64, 0, stream>>>(attn, wl, vp, ctx2);

  gemm_bf16_wmma<<<dim3(DD / 64, (BB * HH) / 16), 32, 0, stream>>>(
      ctx2, Wo_b, bo, 1.0f, out, BB * HH, DD, DD);
}
